// SwinTransformerUNet_16097537426163
// MI455X (gfx1250) — compile-verified, matrix operand order checked by reading, not run on hardware
//
#include <hip/hip_runtime.h>
#include <hip/hip_bf16.h>
#include <math.h>

// ---------------------------------------------------------------------------
// CDNA5 (gfx1250) Swin-UNet forward.
// All linear layers run through a WMMA f32<-f16 tiled GEMM (v_wmma_f32_16x16x32_f16)
// with double-buffered LDS staging.  Window attention (49 tokens, d=32) is a
// VALU kernel with async global->LDS staging (ASYNCcnt tracked); everything
// else is elementwise / gather kernels.
// ---------------------------------------------------------------------------

typedef __attribute__((ext_vector_type(16))) _Float16 v16h;
typedef __attribute__((ext_vector_type(8)))  _Float16 v8h;
typedef __attribute__((ext_vector_type(4)))  _Float16 v4h;
typedef __attribute__((ext_vector_type(8)))  float    v8f;

typedef __attribute__((address_space(1))) int as1_int;
typedef __attribute__((address_space(3))) int as3_int;

union HalfVec { v16h v; v8h h[2]; };

__device__ inline v8f vzero8() {
    v8f v;
#pragma unroll
    for (int i = 0; i < 8; ++i) v[i] = 0.0f;
    return v;
}

// Load a 16x32 f16 fragment from LDS (row-major, stride 32 halves) following the
// CDNA5 WMMA A-matrix striping: lanes 0-15 hold K={0..7,16..23}, lanes 16-31
// hold K={8..15,24..31}; row = base + (lane & 15).  Both halves are contiguous
// 8-half (16B) runs -> two ds_load_b128 per fragment.
__device__ inline v16h frag16(const _Float16* __restrict__ s, int rowBase, int lane) {
    const int r  = rowBase + (lane & 15);
    const int kh = (lane >> 4) << 3;   // 0 or 8
    HalfVec u;
    u.h[0] = *(const v8h*)(s + r * 32 + kh);
    u.h[1] = *(const v8h*)(s + r * 32 + 16 + kh);
    return u.v;
}

// ---------------------------------------------------------------------------
// Tiled WMMA GEMM: C[M,N] = A[M,K] @ B[K,N]  (+bias, +residual, GELU optional)
// A,B,C row-major fp32.  Tile 64x64, BK=32, 128 threads = 4 waves in 2x2,
// each wave owns a 32x32 patch = 2x2 v_wmma_f32_16x16x32_f16 accumulators.
// Double-buffered LDS: next K-tile staged while current tile feeds the WMMAs.
// ---------------------------------------------------------------------------
#define GF_BIAS 1
#define GF_GELU 2
#define GF_RES  4

__global__ __launch_bounds__(128) void k_gemm(
    const float* __restrict__ A, const float* __restrict__ B,
    const float* __restrict__ bias, const float* __restrict__ res,
    float* __restrict__ C, int M, int N, int K, int flags)
{
    __shared__ __align__(16) _Float16 sA[2 * 64 * 32];   // [buf][m][k]
    __shared__ __align__(16) _Float16 sB[2 * 64 * 32];   // [buf][n][k]  (K-transposed)

    const int tid  = threadIdx.x & 127;
    const int lane = tid & 31;
    const int wid  = tid >> 5;           // 0..3
    const int wm   = (wid >> 1) * 32;    // wave row offset inside tile
    const int wn   = (wid & 1) * 32;     // wave col offset inside tile
    const int m0   = blockIdx.y * 64;
    const int n0   = blockIdx.x * 64;

    const bool interior = (m0 + 64 <= M) && (n0 + 64 <= N);

    // Stage one 64x32 A tile and one 32x64 B tile (f32 -> f16) into buffer `buf`.
    auto stageTile = [&](int buf, int kk) {
        _Float16* dA = sA + buf * 2048;
        _Float16* dB = sB + buf * 2048;
        if (interior && (kk + 32 <= K)) {
            // fast path: 4 b128 loads each for A and B, then convert+store
            float4 fa[4], fb[4];
#pragma unroll
            for (int it = 0; it < 4; ++it) {
                const int u = tid + it * 128;
                fa[it] = *(const float4*)(A + (size_t)(m0 + (u >> 3)) * K + (kk + ((u & 7) << 2)));
                fb[it] = *(const float4*)(B + (size_t)(kk + (u >> 4)) * N + (n0 + ((u & 15) << 2)));
            }
#pragma unroll
            for (int it = 0; it < 4; ++it) {
                const int u = tid + it * 128;
                const int r  = u >> 3, k4 = (u & 7) << 2;
                v4h p;
                p[0] = (_Float16)fa[it].x; p[1] = (_Float16)fa[it].y;
                p[2] = (_Float16)fa[it].z; p[3] = (_Float16)fa[it].w;
                *(v4h*)(dA + r * 32 + k4) = p;
                const int kb = u >> 4, n4 = (u & 15) << 2;
                dB[(n4 + 0) * 32 + kb] = (_Float16)fb[it].x;
                dB[(n4 + 1) * 32 + kb] = (_Float16)fb[it].y;
                dB[(n4 + 2) * 32 + kb] = (_Float16)fb[it].z;
                dB[(n4 + 3) * 32 + kb] = (_Float16)fb[it].w;
            }
        } else {
            // slow path: edge tiles, fully guarded
#pragma unroll
            for (int it = 0; it < 16; ++it) {
                const int u = tid + it * 128;
                const int r = u >> 5, k = u & 31;
                const int gm = m0 + r, gk = kk + k;
                const float v = (gm < M && gk < K) ? A[(size_t)gm * K + gk] : 0.0f;
                dA[u] = (_Float16)v;
            }
#pragma unroll
            for (int it = 0; it < 16; ++it) {
                const int u = tid + it * 128;
                const int n = u >> 5, k = u & 31;
                const int gn = n0 + n, gk = kk + k;
                const float v = (gn < N && gk < K) ? B[(size_t)gk * N + gn] : 0.0f;
                dB[n * 32 + k] = (_Float16)v;
            }
        }
    };

    v8f c00 = vzero8(), c01 = vzero8(), c10 = vzero8(), c11 = vzero8();

    stageTile(0, 0);
    for (int k0 = 0; k0 < K; k0 += 32) {
        const int cur = (k0 >> 5) & 1;
        __syncthreads();                       // current buffer staged; prev reads done
        if (k0 + 32 < K) stageTile(cur ^ 1, k0 + 32);   // overlap next stage with math
        // prefetch tile after next (global_prefetch_b8)
        const int kp = k0 + 64;
        if (kp < K) {
            const int gm = m0 + (tid & 63);
            if (gm < M) __builtin_prefetch(&A[(size_t)gm * K + kp], 0, 3);
            const int gn = n0 + (tid & 63);
            if (gn < N) __builtin_prefetch(&B[(size_t)kp * N + gn], 0, 3);
        }

        const _Float16* cA = sA + cur * 2048;
        const _Float16* cB = sB + cur * 2048;
        const v16h a0 = frag16(cA, wm + 0,  lane);
        const v16h a1 = frag16(cA, wm + 16, lane);
        const v16h b0 = frag16(cB, wn + 0,  lane);
        const v16h b1 = frag16(cB, wn + 16, lane);

        c00 = __builtin_amdgcn_wmma_f32_16x16x32_f16(false, a0, false, b0, (short)0, c00, false, false);
        c01 = __builtin_amdgcn_wmma_f32_16x16x32_f16(false, a0, false, b1, (short)0, c01, false, false);
        c10 = __builtin_amdgcn_wmma_f32_16x16x32_f16(false, a1, false, b0, (short)0, c10, false, false);
        c11 = __builtin_amdgcn_wmma_f32_16x16x32_f16(false, a1, false, b1, (short)0, c11, false, false);
    }

    // ---- epilogue: C/D layout -> lanes 0-15: M=r, lanes 16-31: M=8+r, N=lane&15
    const bool hasBias = flags & GF_BIAS;
    const bool doGelu  = flags & GF_GELU;
    const bool hasRes  = flags & GF_RES;
    const int lrow = (lane >> 4) << 3;
    const int lcol = lane & 15;

    v8f* accs[2][2] = { { &c00, &c01 }, { &c10, &c11 } };
#pragma unroll
    for (int mf = 0; mf < 2; ++mf) {
#pragma unroll
        for (int nf = 0; nf < 2; ++nf) {
            v8f& acc = *accs[mf][nf];
#pragma unroll
            for (int r = 0; r < 8; ++r) {
                const int gm = m0 + wm + mf * 16 + lrow + r;
                const int gn = n0 + wn + nf * 16 + lcol;
                if (gm < M && gn < N) {
                    float x = acc[r];
                    if (hasBias) x += bias[gn];
                    if (hasRes)  x += res[(size_t)gm * N + gn];
                    if (doGelu)  x = 0.5f * x * (1.0f + erff(x * 0.7071067811865476f));
                    C[(size_t)gm * N + gn] = x;
                }
            }
        }
    }
}

// ---------------------------------------------------------------------------
// LayerNorm over last dim (one block per token row; in-place safe).
// ---------------------------------------------------------------------------
__global__ __launch_bounds__(256) void k_ln(
    const float* __restrict__ in, float* __restrict__ out,
    const float* __restrict__ g, const float* __restrict__ b, int C)
{
    __shared__ float red[256];
    const size_t t = blockIdx.x;
    const float* row = in + t * C;
    float s = 0.0f;
    for (int i = threadIdx.x; i < C; i += 256) s += row[i];
    red[threadIdx.x] = s; __syncthreads();
    for (int off = 128; off > 0; off >>= 1) {
        if (threadIdx.x < off) red[threadIdx.x] += red[threadIdx.x + off];
        __syncthreads();
    }
    const float mean = red[0] / (float)C;
    __syncthreads();
    float v = 0.0f;
    for (int i = threadIdx.x; i < C; i += 256) { const float d = row[i] - mean; v += d * d; }
    red[threadIdx.x] = v; __syncthreads();
    for (int off = 128; off > 0; off >>= 1) {
        if (threadIdx.x < off) red[threadIdx.x] += red[threadIdx.x + off];
        __syncthreads();
    }
    const float rstd = rsqrtf(red[0] / (float)C + 1e-5f);
    float* orow = out + t * C;
    for (int i = threadIdx.x; i < C; i += 256)
        orow[i] = (row[i] - mean) * rstd * g[i] + b[i];
}

// ---------------------------------------------------------------------------
// Window partition with cyclic shift (roll by -shift), (B,H*W,C) -> (nW*49,C)
// ---------------------------------------------------------------------------
__global__ void k_win_part(const float* __restrict__ in, float* __restrict__ out,
                           int B, int H, int W, int C, int shift)
{
    const size_t idx = (size_t)blockIdx.x * blockDim.x + threadIdx.x;
    const size_t total = (size_t)B * H * W * C;
    if (idx >= total) return;
    const int c = idx % C;
    size_t tok = idx / C;
    const int p = tok % 49; tok /= 49;
    const int nWw = W / 7;
    const int nWimg = (H / 7) * nWw;
    const int b  = tok / nWimg;
    const int wi = tok % nWimg;
    const int wy = wi / nWw, wx = wi % nWw;
    const int py = p / 7,    px = p % 7;
    const int y  = (wy * 7 + py + shift) % H;
    const int xx = (wx * 7 + px + shift) % W;
    out[idx] = in[((size_t)(b * H + y) * W + xx) * C + c];
}

// Window reverse (+shift roll) + residual add: out = shortcut + reverse(win)
__global__ void k_win_rev(const float* __restrict__ win, const float* __restrict__ shortcut,
                          float* __restrict__ out, int B, int H, int W, int C, int shift)
{
    const size_t idx = (size_t)blockIdx.x * blockDim.x + threadIdx.x;
    const size_t total = (size_t)B * H * W * C;
    if (idx >= total) return;
    const int c = idx % C;
    size_t tok = idx / C;
    const int xx = tok % W; tok /= W;
    const int y  = tok % H;
    const int b  = tok / H;
    const int ys = (y - shift + H) % H;
    const int xs = (xx - shift + W) % W;
    const int nWw = W / 7;
    const int nWimg = (H / 7) * nWw;
    const int wy = ys / 7, py = ys % 7, wx = xs / 7, px = xs % 7;
    const size_t widx = ((size_t)((size_t)b * nWimg + wy * nWw + wx) * 49 + py * 7 + px) * C + c;
    out[idx] = win[widx] + shortcut[idx];
}

// ---------------------------------------------------------------------------
// Window attention core: one block = (window, head). 49 tokens, d = 32.
// qkv rows laid out [q(C) | k(C) | v(C)], head h at cols h*32..h*32+31.
// K/V staged to LDS via gfx1250 async global->LDS loads (ASYNCcnt tracked).
// ---------------------------------------------------------------------------
__global__ __launch_bounds__(64) void k_win_attn(
    const float* __restrict__ qkv, const float* __restrict__ rpb,
    float* __restrict__ out, int C, int nh)
{
    __shared__ float kb[49 * 32];
    __shared__ float vb[49 * 32];
    const int w = blockIdx.x, h = blockIdx.y;
    const int t0 = w * 49;
    const int C3 = 3 * C;

#if __has_builtin(__builtin_amdgcn_global_load_async_to_lds_b32)
    for (int u = threadIdx.x; u < 49 * 32; u += 64) {
        const int row = u >> 5, d = u & 31;
        __builtin_amdgcn_global_load_async_to_lds_b32(
            (as1_int*)(qkv + (size_t)(t0 + row) * C3 + C + h * 32 + d),
            (as3_int*)(&kb[u]), 0, 0);
        __builtin_amdgcn_global_load_async_to_lds_b32(
            (as1_int*)(qkv + (size_t)(t0 + row) * C3 + 2 * C + h * 32 + d),
            (as3_int*)(&vb[u]), 0, 0);
    }
#if __has_builtin(__builtin_amdgcn_s_wait_asynccnt)
    __builtin_amdgcn_s_wait_asynccnt(0);
#else
    asm volatile("s_wait_asynccnt 0x0" ::: "memory");
#endif
#else
    for (int u = threadIdx.x; u < 49 * 32; u += 64) {
        const int row = u >> 5, d = u & 31;
        kb[u] = qkv[(size_t)(t0 + row) * C3 + C     + h * 32 + d];
        vb[u] = qkv[(size_t)(t0 + row) * C3 + 2 * C + h * 32 + d];
    }
#endif
    __syncthreads();
    const int i = threadIdx.x;
    if (i >= 49) return;
    float q[32];
    const float scale = 0.17677669529663687f;   // 32^-0.5
    for (int d = 0; d < 32; ++d) q[d] = qkv[(size_t)(t0 + i) * C3 + h * 32 + d] * scale;
    const int yi = i / 7, xi = i % 7;
    float s[49];
    float mx = -1e30f;
    for (int j = 0; j < 49; ++j) {
        float a = 0.0f;
        for (int d = 0; d < 32; ++d) a += q[d] * kb[j * 32 + d];
        const int yj = j / 7, xj = j % 7;
        a += rpb[((yi - yj + 6) * 13 + (xi - xj + 6)) * nh + h];
        s[j] = a;
        if (a > mx) mx = a;
    }
    float sum = 0.0f;
    for (int j = 0; j < 49; ++j) { const float e = __expf(s[j] - mx); s[j] = e; sum += e; }
    const float inv = 1.0f / sum;
    for (int d = 0; d < 32; ++d) {
        float o = 0.0f;
        for (int j = 0; j < 49; ++j) o += s[j] * vb[j * 32 + d];
        out[(size_t)(t0 + i) * C + h * 32 + d] = o * inv;
    }
}

// ---------------------------------------------------------------------------
// Patch merge gather: (B,H,W,C) -> (B,H/2*W/2,4C), quadrant order 00,10,01,11.
// ---------------------------------------------------------------------------
__global__ void k_merge_gather(const float* __restrict__ in, float* __restrict__ out,
                               int B, int H, int W, int C)
{
    const int H2 = H / 2, W2 = W / 2;
    const size_t total = (size_t)B * H2 * W2 * 4 * C;
    const size_t idx = (size_t)blockIdx.x * blockDim.x + threadIdx.x;
    if (idx >= total) return;
    const int c4 = idx % (4 * C);
    size_t tok = idx / (4 * C);
    const int w2 = tok % W2; tok /= W2;
    const int h2 = tok % H2;
    const int b  = tok / H2;
    const int q = c4 / C, c = c4 % C;
    const int dy = q & 1, dx = q >> 1;
    const int y = 2 * h2 + dy, x = 2 * w2 + dx;
    out[idx] = in[((size_t)(b * H + y) * W + x) * C + c];
}

// Patch expand rearrange: in (B,H,W,4*C2) -> out (B,2H,2W,C2)
__global__ void k_expand_rearrange(const float* __restrict__ in, float* __restrict__ out,
                                   int B, int H, int W, int C2)
{
    const size_t total = (size_t)B * 2 * H * 2 * W * C2;
    const size_t idx = (size_t)blockIdx.x * blockDim.x + threadIdx.x;
    if (idx >= total) return;
    const int c = idx % C2;
    size_t tok = idx / C2;
    const int X = tok % (2 * W); tok /= (2 * W);
    const int Y = tok % (2 * H);
    const int b = tok / (2 * H);
    const int h = Y >> 1, i = Y & 1, w = X >> 1, j = X & 1;
    out[idx] = in[((size_t)(b * H + h) * W + w) * (4 * C2) + (i * 2 + j) * C2 + c];
}

// Concat last dim: out(T,C1+C2) = [a(T,C1) | b(T,C2)]
__global__ void k_concat2(const float* __restrict__ a, const float* __restrict__ bsrc,
                          float* __restrict__ out, int T, int C1, int C2)
{
    const size_t total = (size_t)T * (C1 + C2);
    const size_t idx = (size_t)blockIdx.x * blockDim.x + threadIdx.x;
    if (idx >= total) return;
    const int col = idx % (C1 + C2);
    const size_t t = idx / (C1 + C2);
    out[idx] = (col < C1) ? a[t * C1 + col] : bsrc[t * C2 + (col - C1)];
}

// x[b,t,:] += temb[b,:]
__global__ void k_add_temb(float* __restrict__ x, const float* __restrict__ temb,
                           int Ttok, int C)
{
    const size_t idx = (size_t)blockIdx.x * blockDim.x + threadIdx.x;
    const size_t total = (size_t)2 * Ttok * C;
    if (idx >= total) return;
    const int c = idx % C;
    const int b = (int)(idx / ((size_t)Ttok * C));
    x[idx] += temb[b * 96 + c];
}

// ---------------------------------------------------------------------------
// Timestep embedding + MLP (96 -> 384 -> 96), B small.  cos first, then sin.
// ---------------------------------------------------------------------------
__global__ __launch_bounds__(384) void k_time(
    const int* __restrict__ t, const float* __restrict__ w1, const float* __restrict__ b1,
    const float* __restrict__ w2, const float* __restrict__ b2,
    float* __restrict__ temb, int B)
{
    __shared__ float e0[96];
    __shared__ float hid[384];
    for (int b = 0; b < B; ++b) {
        const float tv = (float)t[b];
        if (threadIdx.x < 96) {
            const int i = threadIdx.x;
            const int fi = i % 48;
            const float freq = __expf(-logf(10000.0f) * (float)fi / 48.0f);
            const float ang = tv * freq;
            e0[i] = (i < 48) ? cosf(ang) : sinf(ang);
        }
        __syncthreads();
        {
            const int j = threadIdx.x;
            float s = b1[j];
            for (int i = 0; i < 96; ++i) s += e0[i] * w1[i * 384 + j];
            hid[j] = 0.5f * s * (1.0f + erff(s * 0.7071067811865476f));
        }
        __syncthreads();
        if (threadIdx.x < 96) {
            const int e = threadIdx.x;
            float s = b2[e];
            for (int j = 0; j < 384; ++j) s += hid[j] * w2[j * 96 + e];
            temb[b * 96 + e] = s;
        }
        __syncthreads();
    }
}

// ---------------------------------------------------------------------------
// 1x1 input+cond projection: (B,256,224,224) -> (B,224*224,96)
// One block per pixel, 96 threads over output channels; pixel vectors in LDS.
// ---------------------------------------------------------------------------
__global__ __launch_bounds__(96) void k_proj_in(
    const float* __restrict__ x, const float* __restrict__ cond,
    const float* __restrict__ in_w, const float* __restrict__ in_b,
    const float* __restrict__ cond_w, const float* __restrict__ cond_b,
    float* __restrict__ h0)
{
    __shared__ float px[256];
    __shared__ float pc[256];
    const int pix = blockIdx.x;
    const int b   = blockIdx.y;
    const size_t HW = 224 * 224;
    for (int u = threadIdx.x; u < 256; u += 96) {
        px[u] = x[((size_t)b * 256 + u) * HW + pix];
        pc[u] = cond[((size_t)b * 256 + u) * HW + pix];
    }
    __syncthreads();
    const int e = threadIdx.x;
    float s = in_b[e] + cond_b[e];
    for (int c = 0; c < 256; ++c)
        s += px[c] * in_w[c * 96 + e] + pc[c] * cond_w[c * 96 + e];
    h0[((size_t)b * HW + pix) * 96 + e] = s;
}

// 4x4 stride-4 patch embed: h0 (B,224,224,96) -> tokens (B,3136,96)
__global__ __launch_bounds__(96) void k_patch_embed(
    const float* __restrict__ h0, const float* __restrict__ pw,
    const float* __restrict__ pbias, float* __restrict__ out)
{
    __shared__ float patch[16 * 96];
    const int tokn = blockIdx.x;            // 0..3135
    const int b    = blockIdx.y;
    const int hp = tokn / 56, wp = tokn % 56;
    for (int u = threadIdx.x; u < 1536; u += 96) {
        const int pi = u / 96, c = u % 96;
        const int y = hp * 4 + pi / 4, xx = wp * 4 + (pi & 3);
        patch[u] = h0[((size_t)b * 50176 + y * 224 + xx) * 96 + c];
    }
    __syncthreads();
    const int e = threadIdx.x;
    float s = pbias[e];
    const float* we = pw + (size_t)e * 1536;   // [e][c][i][j]
    for (int c = 0; c < 96; ++c)
        for (int pi = 0; pi < 16; ++pi)
            s += patch[pi * 96 + c] * we[c * 16 + pi];
    out[((size_t)b * 3136 + tokn) * 96 + e] = s;
}

// 4x4 transposed-conv un-embed: tokens (B,3136,96) -> (B,256,224,224)
__global__ void k_unembed(const float* __restrict__ hf, const float* __restrict__ un_w,
                          const float* __restrict__ un_b, float* __restrict__ out, int B)
{
    const size_t total = (size_t)B * 256 * 224 * 224;
    const size_t idx = (size_t)blockIdx.x * blockDim.x + threadIdx.x;
    if (idx >= total) return;
    const int X = idx % 224;
    size_t r = idx / 224;
    const int Y = r % 224; r /= 224;
    const int o = r % 256;
    const int b = (int)(r / 256);
    const int h = Y >> 2, i = Y & 3, w = X >> 2, j = X & 3;
    const float* hrow = hf + ((size_t)b * 3136 + h * 56 + w) * 96;
    float s = un_b[o];
    const int wofs = o * 16 + i * 4 + j;
    for (int e = 0; e < 96; ++e) s += hrow[e] * un_w[(size_t)e * 4096 + wofs];
    out[idx] = s;
}

// ---------------------------------------------------------------------------
// Host-side orchestration
// ---------------------------------------------------------------------------
extern "C" void kernel_launch(void* const* d_in, const int* in_sizes, int n_in,
                              void* d_out, int out_size, void* d_ws, size_t ws_size,
                              hipStream_t stream)
{
    (void)in_sizes; (void)n_in; (void)out_size; (void)ws_size;
    int idx = 0;
    const float* x    = (const float*)d_in[idx++];
    const float* cond = (const float*)d_in[idx++];
    const int*   t    = (const int*)d_in[idx++];
    auto P = [&]() { return (const float*)d_in[idx++]; };

    // --- params in setup_inputs() insertion (DFS) order ---
    const float* in_w = P();   const float* in_b = P();
    const float* cond_w = P(); const float* cond_b = P();
    const float* pe_w = P();   const float* pe_bias = P();
    const float* pe_g = P();   const float* pe_b = P();
    const float* tw1 = P();    const float* tb1 = P();
    const float* tw2 = P();    const float* tb2 = P();

    struct Blk {
        const float *ln1_g, *ln1_b, *qkv_w, *qkv_b, *rpb, *proj_w, *proj_b;
        const float *ln2_g, *ln2_b, *m1_w, *m1_b, *m2_w, *m2_b;
    };
    auto readBlk = [&]() {
        Blk bk;
        bk.ln1_g = P(); bk.ln1_b = P();
        bk.qkv_w = P(); bk.qkv_b = P();
        bk.rpb = P();
        bk.proj_w = P(); bk.proj_b = P();
        bk.ln2_g = P(); bk.ln2_b = P();
        bk.m1_w = P(); bk.m1_b = P();
        bk.m2_w = P(); bk.m2_b = P();
        return bk;
    };

    const int DEPTHS[4] = {2, 2, 6, 2};
    const int HEADSa[4] = {3, 6, 12, 24};

    Blk enc_blocks[4][6];
    const float* merge_g[3]; const float* merge_b[3]; const float* merge_w[3];
    for (int i = 0; i < 4; ++i) {
        for (int j = 0; j < DEPTHS[i]; ++j) enc_blocks[i][j] = readBlk();
        if (i < 3) { merge_g[i] = P(); merge_b[i] = P(); merge_w[i] = P(); }
    }
    Blk bott[2]; bott[0] = readBlk(); bott[1] = readBlk();

    struct Dec { const float *ex_w, *ex_g, *ex_b, *fuse_w, *fuse_b; Blk blocks[6]; int nblk; };
    Dec dec[3];
    const int dec_i[3] = {3, 2, 1};
    for (int k = 0; k < 3; ++k) {
        dec[k].ex_w = P(); dec[k].ex_g = P(); dec[k].ex_b = P();
        dec[k].fuse_w = P(); dec[k].fuse_b = P();
        dec[k].nblk = DEPTHS[dec_i[k] - 1];
        for (int j = 0; j < dec[k].nblk; ++j) dec[k].blocks[j] = readBlk();
    }
    const float* fn_g = P(); const float* fn_b = P();
    const float* un_w = P(); const float* un_b = P();

    // --- workspace carve (floats) ---
    float* W0   = (float*)d_ws;
    float* temb = W0;                               // 192
    float* XA   = W0 + 1024;                        // 650k
    float* XB   = XA + 650000;                      // 650k
    float* skip0 = XB + 650000;                     // 610k
    float* skip1 = skip0 + 610000;                  // 310k
    float* skip2 = skip1 + 310000;                  // 160k
    float* pool  = skip2 + 160000;                  // 4 x 2.5M chunks (10M total)
    const size_t CH = 2500000;
    float* c0 = pool;
    float* c1 = pool + CH;
    float* c2 = pool + 2 * CH;
    float* c3 = pool + 3 * CH;
    float* h0 = pool;                               // 9,633,792 floats, pre-patch-embed only

    const int Bn = 2;

    auto gemm = [&](const float* A, const float* Bm, const float* bias, const float* res,
                    float* Cc, int M, int N, int K, int flags) {
        dim3 g((N + 63) / 64, (M + 63) / 64);
        k_gemm<<<g, 128, 0, stream>>>(A, Bm, bias, res, Cc, M, N, K, flags);
    };
    auto ln = [&](const float* in, float* out, const float* g_, const float* b_, int T, int C) {
        k_ln<<<T, 256, 0, stream>>>(in, out, g_, b_, C);
    };
    auto swin = [&](float*& X, int H, int Wd, int C, int nh, int shift, const Blk& bp) {
        const int T = Bn * H * Wd;
        const int nWimg = (H / 7) * (Wd / 7);
        const int nW = Bn * nWimg;
        const int tot = T * C;
        // attention half
        ln(X, c0, bp.ln1_g, bp.ln1_b, T, C);
        k_win_part<<<(tot + 255) / 256, 256, 0, stream>>>(c0, c1, Bn, H, Wd, C, shift);
        gemm(c1, bp.qkv_w, bp.qkv_b, nullptr, c2, nW * 49, 3 * C, C, GF_BIAS);
        k_win_attn<<<dim3(nW, nh), 64, 0, stream>>>(c2, bp.rpb, c3, C, nh);
        gemm(c3, bp.proj_w, bp.proj_b, nullptr, c1, nW * 49, C, C, GF_BIAS);
        float* Xn = (X == XA) ? XB : XA;
        k_win_rev<<<(tot + 255) / 256, 256, 0, stream>>>(c1, X, Xn, Bn, H, Wd, C, shift);
        X = Xn;
        // MLP half (residual fused into second GEMM epilogue)
        ln(X, c0, bp.ln2_g, bp.ln2_b, T, C);
        gemm(c0, bp.m1_w, bp.m1_b, nullptr, c1, T, 4 * C, C, GF_BIAS | GF_GELU);
        gemm(c1, bp.m2_w, bp.m2_b, X, X, T, C, 4 * C, GF_BIAS | GF_RES);
    };

    // ---- stem ----
    k_time<<<1, 384, 0, stream>>>(t, tw1, tb1, tw2, tb2, temb, Bn);
    k_proj_in<<<dim3(224 * 224, Bn), 96, 0, stream>>>(x, cond, in_w, in_b, cond_w, cond_b, h0);
    k_patch_embed<<<dim3(56 * 56, Bn), 96, 0, stream>>>(h0, pe_w, pe_bias, XA);
    float* X = XA;
    ln(X, X, pe_g, pe_b, Bn * 3136, 96);
    k_add_temb<<<(Bn * 3136 * 96 + 255) / 256, 256, 0, stream>>>(X, temb, 3136, 96);

    // ---- encoder ----
    int H = 56, Wd = 56;
    float* skips[3] = {skip0, skip1, skip2};
    for (int i = 0; i < 4; ++i) {
        const int C = 96 << i;
        for (int j = 0; j < DEPTHS[i]; ++j)
            swin(X, H, Wd, C, HEADSa[i], (j % 2) ? 3 : 0, enc_blocks[i][j]);
        if (i < 3) {
            (void)hipMemcpyAsync(skips[i], X, (size_t)Bn * H * Wd * C * sizeof(float),
                                 hipMemcpyDeviceToDevice, stream);
            const int T2 = Bn * (H / 2) * (Wd / 2);
            const size_t tm = (size_t)T2 * 4 * C;
            k_merge_gather<<<(int)((tm + 255) / 256), 256, 0, stream>>>(X, c0, Bn, H, Wd, C);
            ln(c0, c0, merge_g[i], merge_b[i], T2, 4 * C);
            float* Xn = (X == XA) ? XB : XA;
            gemm(c0, merge_w[i], nullptr, nullptr, Xn, T2, 2 * C, 4 * C, 0);
            X = Xn; H /= 2; Wd /= 2;
        }
    }

    // ---- bottleneck (C=768, 7x7) ----
    for (int j = 0; j < 2; ++j)
        swin(X, H, Wd, 768, 24, (j % 2) ? 3 : 0, bott[j]);

    // ---- decoder ----
    for (int k = 0; k < 3; ++k) {
        const int i = dec_i[k];
        const int C = 96 << i;       // 768, 384, 192
        const int C2 = C / 2;
        const int T = Bn * H * Wd;
        gemm(X, dec[k].ex_w, nullptr, nullptr, c0, T, 2 * C, C, 0);
        float* Xn = (X == XA) ? XB : XA;
        const size_t te = (size_t)T * 2 * C;
        k_expand_rearrange<<<(int)((te + 255) / 256), 256, 0, stream>>>(c0, Xn, Bn, H, Wd, C2);
        X = Xn; H *= 2; Wd *= 2;
        const int T2 = Bn * H * Wd;
        ln(X, X, dec[k].ex_g, dec[k].ex_b, T2, C2);
        const size_t tc = (size_t)T2 * C;
        k_concat2<<<(int)((tc + 255) / 256), 256, 0, stream>>>(X, skips[i - 1], c0, T2, C2, C2);
        Xn = (X == XA) ? XB : XA;
        gemm(c0, dec[k].fuse_w, dec[k].fuse_b, nullptr, Xn, T2, C2, C, GF_BIAS);
        X = Xn;
        for (int j = 0; j < dec[k].nblk; ++j)
            swin(X, H, Wd, C2, HEADSa[i - 1], (j % 2) ? 3 : 0, dec[k].blocks[j]);
    }

    // ---- head ----
    ln(X, X, fn_g, fn_b, Bn * 3136, 96);
    const size_t totOut = (size_t)Bn * 256 * 224 * 224;
    k_unembed<<<(int)((totOut + 255) / 256), 256, 0, stream>>>(X, un_w, un_b, (float*)d_out, Bn);
}
